// LCNChannelStack_4698694222621
// MI455X (gfx1250) — compile-verified, matrix operand order checked by reading, not run on hardware
//
#include <hip/hip_runtime.h>
#include <hip/hip_bf16.h>

#define IN_DIM 14400
#define KNN    25
#define BATCH  256
#define OUTD   2
#define KPAD   1824   // 1800 padded up to a multiple of 32 (57 chunks)

typedef __attribute__((ext_vector_type(16))) _Float16     v16h;
typedef __attribute__((ext_vector_type(8)))  float        v8f;
typedef __attribute__((ext_vector_type(8)))  unsigned int v8u;

// ---------------------------------------------------------------------------
// Transpose x (256 x 28800, row-major) -> xT (28800 x 256) so that gathers
// become coalesced (lanes = batch) with uniform per-neuron addresses.
// ---------------------------------------------------------------------------
__global__ void transpose_kernel(const float* __restrict__ x, float* __restrict__ xT) {
  __shared__ float tile[32][33];
  const int f0 = blockIdx.x * 32;
  const int b0 = blockIdx.y * 32;
  const int tx = threadIdx.x, ty = threadIdx.y;
#pragma unroll
  for (int i = 0; i < 4; ++i) {
    int b = b0 + ty + i * 8;
    tile[ty + i * 8][tx] = x[(size_t)b * (2 * IN_DIM) + f0 + tx];
  }
  __syncthreads();
#pragma unroll
  for (int i = 0; i < 4; ++i) {
    int f = f0 + ty + i * 8;
    xT[(size_t)f * BATCH + b0 + tx] = tile[tx][ty + i * 8];
  }
}

// ---------------------------------------------------------------------------
// One LCN layer, batch-major: out[d,b] = relu(sum_k w[d,k]*srcT[knn[d,k], b]).
// 4 neurons per block; indices/weights staged in LDS (uniform across lanes),
// every gathered load is a fully coalesced 128B line per wave.
// ---------------------------------------------------------------------------
__global__ void lcn_layer_kernel(const float* __restrict__ srcT,
                                 const int*   __restrict__ knn,
                                 const float* __restrict__ w,
                                 const float* __restrict__ bias,
                                 float*       __restrict__ dstT) {
  __shared__ int   sidx[4 * KNN];
  __shared__ float sw  [4 * KNN];
  const int d0 = blockIdx.x * 4;
  const int t  = threadIdx.x;           // 256 threads = batch
  if (t < 4 * KNN) {
    sidx[t] = knn[(size_t)d0 * KNN + t];
    sw[t]   = w  [(size_t)d0 * KNN + t];
  }
  __syncthreads();
  const int b = t;
#pragma unroll
  for (int dd = 0; dd < 4; ++dd) {
    float acc = 0.f;
#pragma unroll
    for (int k = 0; k < KNN; ++k) {
      acc = fmaf(sw[dd * KNN + k],
                 srcT[(size_t)sidx[dd * KNN + k] * BATCH + b], acc);
    }
    float o = acc + bias[d0 + dd];
    dstT[(size_t)(d0 + dd) * BATCH + b] = o > 0.f ? o : 0.f;
  }
}

// ---------------------------------------------------------------------------
// Zero the pad rows (1800..1823) of both channels' padded a2 buffers.
// 2 * 24 * 256 = 12288 elements, grid 48 x 256.
// ---------------------------------------------------------------------------
__global__ void zero_pad_kernel(float* __restrict__ a2) {
  int i  = blockIdx.x * 256 + threadIdx.x;     // 0 .. 12287
  int ch = i / (24 * BATCH);
  int r  = i % (24 * BATCH);
  a2[(size_t)ch * KPAD * BATCH + (size_t)1800 * BATCH + r] = 0.f;
}

// ---------------------------------------------------------------------------
// Expand fc weights (2 x 1800 f32) into padded f16 B-panels (2ch x 16 x 1824),
// zeros for n >= 2 and f >= 1800.  2*16*1824 = 58368 elems, grid 228 x 256.
// ---------------------------------------------------------------------------
__global__ void prep_b_kernel(const float* __restrict__ fwp,
                              const float* __restrict__ fwn,
                              _Float16*    __restrict__ Bp) {
  int i   = blockIdx.x * 256 + threadIdx.x;    // 0 .. 58367
  int ch  = i / (16 * KPAD);
  int rem = i % (16 * KPAD);
  int n   = rem / KPAD;
  int f   = rem % KPAD;
  const float* fw = ch ? fwn : fwp;
  float v = (n < OUTD && f < 1800) ? fw[n * 1800 + f] : 0.f;
  Bp[i] = (_Float16)v;
}

// ---------------------------------------------------------------------------
// Final per-channel FC (256x1800)*(1800x2) via v_wmma_f32_16x16x32_f16.
// One wave per 16-batch tile; K padded to 1824 so the loop has NO guards:
// A fragment: 16 unconditional f32 loads + cvt;  B fragment: 8 dword loads
// from the pre-expanded f16 panel, bit-cast straight into the v16h operand.
// A layout (16-bit 16x32): lanes<16 hold M=lane, K in {0..7,16..23};
// lanes>=16 hold M=lane-16, K shifted by +8.  B mirrors with N=lane&15.
// C/D layout: n = lane&15, m = r + 8*(lane>=16).
// ---------------------------------------------------------------------------
__global__ void fc_wmma_kernel(const float*    __restrict__ actT,  // (1824 x 256)
                               const _Float16* __restrict__ Bp,    // (16 x 1824)
                               const float*    __restrict__ fb,    // (2)
                               float*          __restrict__ hbuf,  // (256 x 4)
                               int col_off) {
  const int lane  = threadIdx.x;        // 32 threads = one wave
  const int n     = lane & 15;          // == m for the A side
  const int khalf = (lane >> 4) ? 8 : 0;
  const int bbase = blockIdx.x * 16;
  const float* ac = actT + bbase + n;                       // column base
  const unsigned int* bp32 = (const unsigned int*)(Bp + (size_t)n * KPAD);
  v8f c = {};
  for (int f0 = 0; f0 < KPAD; f0 += 32) {
    // B fragment: 8 dwords, each dword = two consecutive f16 K-values.
    v8u bu;
#pragma unroll
    for (int j = 0; j < 8; ++j) {
      int fb2 = f0 + khalf + ((j < 4) ? 2 * j : 16 + 2 * (j - 4));
      bu[j] = bp32[fb2 >> 1];
    }
    v16h bm = __builtin_bit_cast(v16h, bu);
    // A fragment: 16 unconditional f32 loads (half-wave coalesced) + cvt.
    v16h a;
#pragma unroll
    for (int j = 0; j < 8; ++j) {
#pragma unroll
      for (int tt = 0; tt < 2; ++tt) {
        int f = f0 + khalf + ((j < 4) ? (2 * j + tt) : (16 + 2 * (j - 4) + tt));
        a[2 * j + tt] = (_Float16)ac[(size_t)f * BATCH];
      }
    }
    c = __builtin_amdgcn_wmma_f32_16x16x32_f16(false, a, false, bm,
                                               (short)0, c, false, false);
  }
  if (n < OUTD) {
    const int mofs = (lane >> 4) ? 8 : 0;
#pragma unroll
    for (int r = 0; r < 8; ++r) {
      int bg = bbase + r + mofs;
      hbuf[(size_t)bg * 4 + col_off + n] = c[r] + fb[n];
    }
  }
}

// ---------------------------------------------------------------------------
// h = relu(concat(xpos, xneg));  out = h @ fc3_w.T + fc3_b   (256 x 2)
// ---------------------------------------------------------------------------
__global__ void fc3_kernel(const float* __restrict__ hbuf,
                           const float* __restrict__ w3,  // (2 x 4)
                           const float* __restrict__ b3,  // (2)
                           float*       __restrict__ out) {
  const int b = threadIdx.x;
  float h[4];
#pragma unroll
  for (int j = 0; j < 4; ++j) {
    float v = hbuf[b * 4 + j];
    h[j] = v > 0.f ? v : 0.f;
  }
#pragma unroll
  for (int o = 0; o < 2; ++o) {
    float acc = b3[o];
#pragma unroll
    for (int j = 0; j < 4; ++j) acc = fmaf(w3[o * 4 + j], h[j], acc);
    out[b * 2 + o] = acc;
  }
}

extern "C" void kernel_launch(void* const* d_in, const int* in_sizes, int n_in,
                              void* d_out, int out_size, void* d_ws, size_t ws_size,
                              hipStream_t stream) {
  (void)in_sizes; (void)n_in; (void)out_size; (void)ws_size;
  const float* x     = (const float*)d_in[0];
  const int*   knn0  = (const int*)d_in[1];
  const int*   knn1  = (const int*)d_in[2];
  const int*   knn2  = (const int*)d_in[3];
  const float* wp[3] = {(const float*)d_in[4], (const float*)d_in[6], (const float*)d_in[8]};
  const float* bp[3] = {(const float*)d_in[5], (const float*)d_in[7], (const float*)d_in[9]};
  const float* fcp_w = (const float*)d_in[10];
  const float* fcp_b = (const float*)d_in[11];
  const float* wn[3] = {(const float*)d_in[12], (const float*)d_in[14], (const float*)d_in[16]};
  const float* bn[3] = {(const float*)d_in[13], (const float*)d_in[15], (const float*)d_in[17]};
  const float* fcn_w = (const float*)d_in[18];
  const float* fcn_b = (const float*)d_in[19];
  const float* fc3_w = (const float*)d_in[20];
  const float* fc3_b = (const float*)d_in[21];

  float* ws = (float*)d_ws;
  float* xT = ws;                               // 28800*256
  float* a0 = xT + (size_t)28800 * BATCH;       // 2 * 7200*256
  float* a1 = a0 + (size_t)2 * 7200 * BATCH;    // 2 * 3600*256
  float* a2 = a1 + (size_t)2 * 3600 * BATCH;    // 2 * 1824*256 (padded)
  float* bpan_f = a2 + (size_t)2 * KPAD * BATCH;
  _Float16* Bp = (_Float16*)bpan_f;             // 2 * 16 * 1824 halves
  float* hb = bpan_f + (size_t)(2 * 16 * KPAD) / 2 + 16;  // 256*4

  transpose_kernel<<<dim3(900, 8), dim3(32, 8), 0, stream>>>(x, xT);
  zero_pad_kernel<<<48, 256, 0, stream>>>(a2);
  prep_b_kernel<<<228, 256, 0, stream>>>(fcp_w, fcn_w, Bp);

  const int* knns[3] = {knn0, knn1, knn2};
  const int  dims[3] = {7200, 3600, 1800};
  for (int ch = 0; ch < 2; ++ch) {
    const float* const* wch = ch ? wn : wp;
    const float* const* bch = ch ? bn : bp;
    const float* src = xT + (size_t)ch * IN_DIM * BATCH;
    float* dsts[3] = {a0 + (size_t)ch * 7200 * BATCH,
                      a1 + (size_t)ch * 3600 * BATCH,
                      a2 + (size_t)ch * KPAD * BATCH};
    for (int L = 0; L < 3; ++L) {
      lcn_layer_kernel<<<dims[L] / 4, 256, 0, stream>>>(src, knns[L], wch[L], bch[L], dsts[L]);
      src = dsts[L];
    }
    fc_wmma_kernel<<<16, 32, 0, stream>>>(dsts[2], Bp + (size_t)ch * 16 * KPAD,
                                          ch ? fcn_b : fcp_b, hb, 2 * ch);
  }
  fc3_kernel<<<1, 256, 0, stream>>>(hb, fc3_w, fc3_b, (float*)d_out);
}
